// InfoNCECauchyBatchAware_47528108098306
// MI455X (gfx1250) — matrix operand order repaired and checked by reading, not compiled, not run on hardware
//
#include <hip/hip_runtime.h>
#include <math.h>

typedef __attribute__((ext_vector_type(8)))  float        v8f;
typedef __attribute__((ext_vector_type(16))) __bf16       v16bf;
typedef __attribute__((ext_vector_type(4)))  unsigned int v4u;
typedef __attribute__((ext_vector_type(8)))  int          v8i;
typedef __attribute__((ext_vector_type(4)))  int          v4i;

#define BSZ   8192
#define FDIM  128
#define KC    32
#define LDPS  40   // LDS row stride in bf16 elems = 20 dwords (16 data + 4 pad, matches TDM pad)

__device__ __forceinline__ v8f wmma_bf16_k32(v16bf a, v16bf b, v8f c) {
  // D = A(16x32,bf16) * B(32x16,bf16) + C(f32)
  return __builtin_amdgcn_wmma_f32_16x16x32_bf16(
      /*neg_a=*/false, a, /*neg_b=*/false, b,
      /*c_mod=*/(short)0, c, /*reuse_a=*/false, /*reuse_b=*/false);
}

__device__ __forceinline__ unsigned short f2bf(float f) {   // round-to-nearest-even
  unsigned u = __float_as_uint(f);
  unsigned r = (u + 0x7fffu + ((u >> 16) & 1u)) >> 16;
  return (unsigned short)r;
}
__device__ __forceinline__ float bf2f(unsigned short h) {
  return __uint_as_float(((unsigned)h) << 16);
}

typedef union { v16bf v; unsigned u[8]; } frag_t;

// ---- TDM: one 2D bf16 tile (tile_d0 elems x tile_d1 rows) global -> LDS, padded ----
// D# layout per CDNA5 ISA (group0 128b, group1 256b).
__device__ __forceinline__ void tdm_load_tile_bf16(const unsigned short* gsrc,
                                                   unsigned lds_byte_addr,
                                                   unsigned tile_d0, unsigned tile_d1,
                                                   unsigned tensor_d0, unsigned row_stride) {
#if __has_builtin(__builtin_amdgcn_tensor_load_to_lds)
  unsigned long long ga = (unsigned long long)(const void*)gsrc;
  v4u g0;
  g0[0] = 1u;                                        // count=1, user-mode, no gather
  g0[1] = lds_byte_addr;                             // lds_addr (bytes)        [63:32]
  g0[2] = (unsigned)(ga & 0xffffffffull);            // global_addr lo          [95:64]
  g0[3] = (unsigned)((ga >> 32) & 0x01ffffffull)     // global_addr hi (57-bit) [120:96]
        | (2u << 30);                                // type = 2 ("image")      [127:126]
  v8i g1;
  // data_size=1 (2B), pad_enable, pad_interval=3 (16 dwords), pad_amount=3 (4 dwords)
  g1[0] = (int)((1u << 16) | (1u << 20) | (3u << 22) | (3u << 25));
  g1[1] = (int)((tensor_d0 & 0xffffu) << 16);        // tensor_dim0 lo16 @ [63:48]
  g1[2] = (int)((tensor_d0 >> 16) | ((0x100000u & 0xffffu) << 16)); // td0 hi | tensor_dim1 lo16
  g1[3] = (int)((0x100000u >> 16) | (tile_d0 << 16));               // td1 hi | tile_dim0
  g1[4] = (int)(tile_d1 & 0xffffu);                  // tile_dim1 ; tile_dim2 = 0
  g1[5] = (int)row_stride;                           // tensor_dim0_stride lo32
  g1[6] = 0;                                         // stride hi16 ; dim1_stride lo16
  g1[7] = 0;
  v4i z4 = (v4i)0;
#if defined(__clang_major__) && (__clang_major__ >= 23)
  v8i z8 = (v8i)0;
  __builtin_amdgcn_tensor_load_to_lds(g0, g1, z4, z4, z8, 0);
#else
  __builtin_amdgcn_tensor_load_to_lds(g0, g1, z4, z4, 0);
#endif
#endif
}

__device__ __forceinline__ unsigned lds_offset_of(const void* p) {
  // generic shared-aperture pointer keeps the LDS byte offset in addr[31:0]
  return (unsigned)(unsigned long long)p;
}

// ---------------- kernel 0: zero global accumulators ----------------
__global__ void zero_kernel(float* __restrict__ p, int n) {
  int i = blockIdx.x * blockDim.x + threadIdx.x;
  if (i < n) p[i] = 0.0f;
}

// ---------------- kernel 0b: one-time f32 -> bf16 hi/lo split ----------------
__global__ void presplit_kernel(const float* __restrict__ feat,
                                unsigned short* __restrict__ hi,
                                unsigned short* __restrict__ lo) {
  int i = blockIdx.x * blockDim.x + threadIdx.x;   // one float4 per thread
  const float4 x = ((const float4*)feat)[i];
  unsigned short h0 = f2bf(x.x), h1 = f2bf(x.y), h2 = f2bf(x.z), h3 = f2bf(x.w);
  unsigned short l0 = f2bf(x.x - bf2f(h0)), l1 = f2bf(x.y - bf2f(h1));
  unsigned short l2 = f2bf(x.z - bf2f(h2)), l3 = f2bf(x.w - bf2f(h3));
  uint2 ph, pl;
  ph.x = (unsigned)h0 | ((unsigned)h1 << 16); ph.y = (unsigned)h2 | ((unsigned)h3 << 16);
  pl.x = (unsigned)l0 | ((unsigned)l1 << 16); pl.y = (unsigned)l2 | ((unsigned)l3 << 16);
  ((uint2*)hi)[i] = ph;
  ((uint2*)lo)[i] = pl;
}

// ---------------- kernel 1: row norms + diagonal term (full f32) ----------------
__global__ void norms_kernel(const float* __restrict__ feat,
                             float* __restrict__ na, float* __restrict__ nb,
                             float* __restrict__ diag_s) {
  int r = blockIdx.x * blockDim.x + threadIdx.x;
  if (r >= BSZ) return;
  const float4* arow = (const float4*)(feat + (size_t)r * FDIM);
  const float4* brow = (const float4*)(feat + (size_t)(r + BSZ) * FDIM);
  float sa = 0.f, sb = 0.f, sab = 0.f;
#pragma unroll 8
  for (int k = 0; k < FDIM / 4; ++k) {
    float4 x = arow[k], y = brow[k];
    sa  += x.x * x.x + x.y * x.y + x.z * x.z + x.w * x.w;
    sb  += y.x * y.x + y.y * y.y + y.z * y.z + y.w * y.w;
    sab += x.x * y.x + x.y * y.y + x.z * y.z + x.w * y.w;
  }
  na[r] = sa;
  nb[r] = sb;
  float d2 = fmaxf(sa + sb - 2.0f * sab, 0.0f);
  diag_s[r] = 0.5f / (d2 + 1.0f);   // W[i,i] = BATCH_WEIGHT = 0.5
}

// ---------------- kernel 2: tiled bf16x3 WMMA similarity + reductions ----------------
// PRE=true : TDM (tensor_load_to_lds) streams pre-split bf16 hi/lo tiles into LDS
// PRE=false: split f32 -> bf16 hi/lo while staging manually (workspace fallback)
template <bool PRE>
__global__ __launch_bounds__(256)
void sim_tile_kernel(const float* __restrict__ feat,
                     const unsigned short* __restrict__ featHi,
                     const unsigned short* __restrict__ featLo,
                     const int*   __restrict__ labels,
                     const float* __restrict__ na, const float* __restrict__ nb,
                     float* __restrict__ accAA, float* __restrict__ accBB,
                     float* __restrict__ accABr, float* __restrict__ accABc) {
  __shared__ unsigned short sXhi[128 * LDPS];
  __shared__ unsigned short sXlo[128 * LDPS];
  __shared__ unsigned short sYhi[128 * LDPS];
  __shared__ unsigned short sYlo[128 * LDPS];
  __shared__ float s_nx[128], s_ny[128];
  __shared__ int   s_li[128], s_lj[128];
  __shared__ float s_rowsum[128], s_colsum[128];

  const int z  = blockIdx.z;            // 0=AA 1=BB 2=AB
  const int i0 = blockIdx.y * 128;
  const int j0 = blockIdx.x * 128;

  const size_t boff = (size_t)BSZ * FDIM;
  const float* Xg = (z == 1) ? feat + boff : feat;
  const float* Yg = (z == 0) ? feat : feat + boff;
  const unsigned short* XgHi = (z == 1) ? featHi + boff : featHi;
  const unsigned short* XgLo = (z == 1) ? featLo + boff : featLo;
  const unsigned short* YgHi = (z == 0) ? featHi : featHi + boff;
  const unsigned short* YgLo = (z == 0) ? featLo : featLo + boff;
  const float* nxp = (z == 1) ? nb : na;
  const float* nyp = (z == 0) ? na : nb;
  float* rowAcc = (z == 0) ? accAA : (z == 1) ? accBB : accABr;
  float* colAcc = (z == 2) ? accABc : nullptr;
  const bool zero_diag = (z != 2);

  const int tid  = threadIdx.x;
  const int lane = tid & 31;
  const int wave = tid >> 5;
  const int wy = wave >> 1;             // 0..3 -> 32-row strip
  const int wx = wave & 1;              // 0..1 -> 64-col strip
  const int rbase = wy * 32;
  const int cbase = wx * 64;

  if (tid < 128) {
    s_nx[tid] = nxp[i0 + tid];
    s_li[tid] = labels[i0 + tid];
    s_rowsum[tid] = 0.0f;
  } else {
    int t = tid - 128;
    s_ny[t] = nyp[j0 + t];
    s_lj[t] = labels[j0 + t];
    s_colsum[t] = 0.0f;
  }

  v8f acc[2][4];
#pragma unroll
  for (int r = 0; r < 2; ++r)
#pragma unroll
    for (int c = 0; c < 4; ++c) acc[r][c] = (v8f){};

  const int ml   = lane & 15;
  const int hsel = lane >> 4;
  const int row  = tid >> 1;            // manual staging assignment
  const int colb = (tid & 1) * 16;

  for (int kc = 0; kc < FDIM; kc += KC) {
    __syncthreads();   // previous chunk fully consumed (also covers staging above)
    if (PRE) {
      // Tensor Data Mover: 4 descriptors per chunk, issued once by wave 0 (EXEC ignored)
      if (wave == 0) {
        const size_t ofsX = (size_t)i0 * FDIM + kc;
        const size_t ofsY = (size_t)j0 * FDIM + kc;
        tdm_load_tile_bf16(XgHi + ofsX, lds_offset_of(sXhi), KC, 128, FDIM, FDIM);
        tdm_load_tile_bf16(XgLo + ofsX, lds_offset_of(sXlo), KC, 128, FDIM, FDIM);
        tdm_load_tile_bf16(YgHi + ofsY, lds_offset_of(sYhi), KC, 128, FDIM, FDIM);
        tdm_load_tile_bf16(YgLo + ofsY, lds_offset_of(sYlo), KC, 128, FDIM, FDIM);
        __builtin_amdgcn_s_wait_tensorcnt(0);   // DMA complete before barrier release
      }
    } else {
      unsigned* dxh = (unsigned*)&sXhi[row * LDPS + colb];
      unsigned* dxl = (unsigned*)&sXlo[row * LDPS + colb];
      unsigned* dyh = (unsigned*)&sYhi[row * LDPS + colb];
      unsigned* dyl = (unsigned*)&sYlo[row * LDPS + colb];
      const float4* srcx = (const float4*)(Xg + (size_t)(i0 + row) * FDIM + kc + colb);
      const float4* srcy = (const float4*)(Yg + (size_t)(j0 + row) * FDIM + kc + colb);
#pragma unroll
      for (int q = 0; q < 4; ++q) {
        float4 x = srcx[q];
        unsigned short h0 = f2bf(x.x), h1 = f2bf(x.y), h2 = f2bf(x.z), h3 = f2bf(x.w);
        unsigned short l0 = f2bf(x.x - bf2f(h0)), l1 = f2bf(x.y - bf2f(h1));
        unsigned short l2 = f2bf(x.z - bf2f(h2)), l3 = f2bf(x.w - bf2f(h3));
        dxh[q * 2 + 0] = (unsigned)h0 | ((unsigned)h1 << 16);
        dxh[q * 2 + 1] = (unsigned)h2 | ((unsigned)h3 << 16);
        dxl[q * 2 + 0] = (unsigned)l0 | ((unsigned)l1 << 16);
        dxl[q * 2 + 1] = (unsigned)l2 | ((unsigned)l3 << 16);
        float4 y = srcy[q];
        h0 = f2bf(y.x); h1 = f2bf(y.y); h2 = f2bf(y.z); h3 = f2bf(y.w);
        l0 = f2bf(y.x - bf2f(h0)); l1 = f2bf(y.y - bf2f(h1));
        l2 = f2bf(y.z - bf2f(h2)); l3 = f2bf(y.w - bf2f(h3));
        dyh[q * 2 + 0] = (unsigned)h0 | ((unsigned)h1 << 16);
        dyh[q * 2 + 1] = (unsigned)h2 | ((unsigned)h3 << 16);
        dyl[q * 2 + 0] = (unsigned)l0 | ((unsigned)l1 << 16);
        dyl[q * 2 + 1] = (unsigned)l2 | ((unsigned)l3 << 16);
      }
    }
    __syncthreads();

    // ---- one K=32 step: dot += Xhi*Yhi + Xlo*Yhi + Xhi*Ylo ----
    frag_t axh[2], axl[2], byh[4], byl[4];
    // A-fragment K map: k(v) = 2*(v&3) + 8*hsel + 16*(v>>2)    (16-bit A 16x32 layout)
    // B-fragment K map: k(v) = 2*(v&3) + 8*hsel + 16*(v>>2)    (same K order as A)
#pragma unroll
    for (int r = 0; r < 2; ++r) {
      const unsigned short* base = &sXhi[(rbase + r * 16 + ml) * LDPS];
#pragma unroll
      for (int v = 0; v < 8; ++v)
        axh[r].u[v] = *(const unsigned*)&base[((v & 3) << 1) + (hsel << 3) + ((v >> 2) << 4)];
    }
#pragma unroll
    for (int c = 0; c < 4; ++c) {
      const unsigned short* base = &sYhi[(cbase + c * 16 + ml) * LDPS];
#pragma unroll
      for (int v = 0; v < 8; ++v)
        byh[c].u[v] = *(const unsigned*)&base[((v & 3) << 1) + (hsel << 3) + ((v >> 2) << 4)];
    }
#pragma unroll
    for (int r = 0; r < 2; ++r)
#pragma unroll
      for (int c = 0; c < 4; ++c)
        acc[r][c] = wmma_bf16_k32(axh[r].v, byh[c].v, acc[r][c]);   // hi*hi

#pragma unroll
    for (int r = 0; r < 2; ++r) {
      const unsigned short* base = &sXlo[(rbase + r * 16 + ml) * LDPS];
#pragma unroll
      for (int v = 0; v < 8; ++v)
        axl[r].u[v] = *(const unsigned*)&base[((v & 3) << 1) + (hsel << 3) + ((v >> 2) << 4)];
    }
#pragma unroll
    for (int r = 0; r < 2; ++r)
#pragma unroll
      for (int c = 0; c < 4; ++c)
        acc[r][c] = wmma_bf16_k32(axl[r].v, byh[c].v, acc[r][c]);   // lo*hi

#pragma unroll
    for (int c = 0; c < 4; ++c) {
      const unsigned short* base = &sYlo[(cbase + c * 16 + ml) * LDPS];
#pragma unroll
      for (int v = 0; v < 8; ++v)
        byl[c].u[v] = *(const unsigned*)&base[((v & 3) << 1) + (hsel << 3) + ((v >> 2) << 4)];
    }
#pragma unroll
    for (int r = 0; r < 2; ++r)
#pragma unroll
      for (int c = 0; c < 4; ++c)
        acc[r][c] = wmma_bf16_k32(axh[r].v, byl[c].v, acc[r][c]);   // hi*lo
  }
  __syncthreads();

  // ---- epilogue: cauchy transform + weight, register-accumulated reductions ----
  const int n    = lane & 15;
  const int half = lane >> 4;
  const int mofs = half * 8;

  float rs[2][8];
  float cs[4] = {0.f, 0.f, 0.f, 0.f};
#pragma unroll
  for (int r = 0; r < 2; ++r)
#pragma unroll
    for (int v = 0; v < 8; ++v) rs[r][v] = 0.f;

#pragma unroll
  for (int r = 0; r < 2; ++r) {
#pragma unroll
    for (int c = 0; c < 4; ++c) {
      const int tj = cbase + c * 16 + n;
      const int gj = j0 + tj;
      const float nyj = s_ny[tj];
      const int   lj  = s_lj[tj];
#pragma unroll
      for (int v = 0; v < 8; ++v) {
        const int ti = rbase + r * 16 + mofs + v;
        const int gi = i0 + ti;
        float d2 = fmaxf(s_nx[ti] + nyj - 2.0f * acc[r][c][v], 0.0f);
        float s  = 1.0f / (d2 + 1.0f);
        if (s_li[ti] == lj) s *= 0.5f;
        if (zero_diag && gi == gj) s = 0.0f;
        rs[r][v] += s;
        cs[c]    += s;
      }
    }
  }
  if (colAcc) {
#pragma unroll
    for (int c = 0; c < 4; ++c) {
      float t = cs[c] + __shfl_xor(cs[c], 16, 32);
      if (half == 0) atomicAdd(&s_colsum[cbase + c * 16 + n], t);
    }
  }
#pragma unroll
  for (int r = 0; r < 2; ++r) {
#pragma unroll
    for (int v = 0; v < 8; ++v) {
      float t = rs[r][v];
      t += __shfl_xor(t, 1, 32);
      t += __shfl_xor(t, 2, 32);
      t += __shfl_xor(t, 4, 32);
      t += __shfl_xor(t, 8, 32);
      if (n == 0) atomicAdd(&s_rowsum[rbase + r * 16 + mofs + v], t);
    }
  }
  __syncthreads();

  if (tid < 128) {
    atomicAdd(&rowAcc[i0 + tid], s_rowsum[tid]);
    if (colAcc) atomicAdd(&colAcc[j0 + tid], s_colsum[tid]);
  }
}

// ---------------- kernel 3: finalize scalar loss ----------------
__global__ void finalize_kernel(const float* __restrict__ diag_s,
                                const float* __restrict__ accAA,
                                const float* __restrict__ accBB,
                                const float* __restrict__ accABr,
                                const float* __restrict__ accABc,
                                float* __restrict__ out) {
  __shared__ float red[256];
  float t = 0.0f;
  for (int i = threadIdx.x; i < BSZ; i += blockDim.x) {
    float al = logf(diag_s[i]);                 // tempered alignment term
    float l1 = logf(accABc[i] + accBB[i]);      // logsumexp_1 term
    float l2 = logf(accAA[i] + accABr[i]);      // logsumexp_2 term
    t += al - 0.5f * (l1 + l2);
  }
  red[threadIdx.x] = t;
  __syncthreads();
  for (int s = 128; s > 0; s >>= 1) {
    if (threadIdx.x < s) red[threadIdx.x] += red[threadIdx.x + s];
    __syncthreads();
  }
  if (threadIdx.x == 0) out[0] = -red[0] / (float)BSZ;
}

extern "C" void kernel_launch(void* const* d_in, const int* in_sizes, int n_in,
                              void* d_out, int out_size, void* d_ws, size_t ws_size,
                              hipStream_t stream) {
  const float* feat   = (const float*)d_in[0];   // (2B, F) f32
  const int*   labels = (const int*)d_in[1];     // (B,) int
  float* out = (float*)d_out;
  float* ws  = (float*)d_ws;

  float* na     = ws;                 // [B]
  float* nb     = ws + 8192;          // [B]
  float* diag_s = ws + 16384;         // [B]
  float* accAA  = ws + 24576;         // [B]
  float* accBB  = ws + 32768;         // [B]
  float* accABr = ws + 40960;         // [B]
  float* accABc = ws + 49152;         // [B]

  const size_t scalar_bytes = (size_t)57344 * sizeof(float);
  const size_t nfeat = (size_t)2 * BSZ * FDIM;               // 2M elements
  unsigned short* featHi = (unsigned short*)((char*)d_ws + scalar_bytes);
  unsigned short* featLo = featHi + nfeat;
  const size_t need = scalar_bytes + 2 * nfeat * sizeof(unsigned short);
  const bool pre = (ws_size >= need);

  zero_kernel<<<(4 * BSZ + 255) / 256, 256, 0, stream>>>(accAA, 4 * BSZ);
  norms_kernel<<<BSZ / 256, 256, 0, stream>>>(feat, na, nb, diag_s);

  dim3 grid(BSZ / 128, BSZ / 128, 3);
  if (pre) {
    presplit_kernel<<<(int)(nfeat / 4 / 256), 256, 0, stream>>>(feat, featHi, featLo);
    sim_tile_kernel<true><<<grid, 256, 0, stream>>>(feat, featHi, featLo, labels, na, nb,
                                                    accAA, accBB, accABr, accABc);
  } else {
    sim_tile_kernel<false><<<grid, 256, 0, stream>>>(feat, featHi, featLo, labels, na, nb,
                                                     accAA, accBB, accABr, accABc);
  }

  finalize_kernel<<<1, 256, 0, stream>>>(diag_s, accAA, accBB, accABr, accABc, out);
}
